// Matching_1331439862522
// MI455X (gfx1250) — matrix operand order repaired and checked by reading, not compiled, not run on hardware
//
#include <hip/hip_runtime.h>

// ---------------------------------------------------------------------------
// MI455X (gfx1250) implementation of the PWC-Net "matching" block.
// Conv stack (168 GFLOP) = implicit GEMM on v_wmma_f32_16x16x32_f16 with
// TDM (tensor_load_to_lds) staging of the activation halo patch into LDS.
// Warp / correlation / upsample are memory-bound f32 kernels.
// ---------------------------------------------------------------------------

typedef __attribute__((ext_vector_type(16))) _Float16 v16h;
typedef __attribute__((ext_vector_type(8)))  _Float16 v8h;
typedef __attribute__((ext_vector_type(8)))  float    v8f;
typedef __attribute__((ext_vector_type(4)))  unsigned int u32x4;
typedef __attribute__((ext_vector_type(8)))  int i32x8;
typedef __attribute__((ext_vector_type(4)))  int i32x4;

constexpr int B = 4, C = 64, H = 192, W = 256;
constexpr int Hh = H / 2, Wh = W / 2;
constexpr int Hp = H + 2, Wp = W + 2;          // 1-px zero halo for 3x3 convs
constexpr int PR = 4;                          // corr radius -> 9x9 = 81 taps
constexpr float DIV_FLOW = 1.0f / 1.25f;       // WARP_DIV
constexpr int HALO = 2 * Wp + 2 * (Hp - 2);    // 900 border pixels
constexpr int MT = 2;                          // pixel tiles (16 px) per block

// ---------------------------------------------------------------------------
// 1) Transposed-conv 2x flow upsample (groups=2, 4x4 kernel, flipped weights)
// ---------------------------------------------------------------------------
__global__ void upflow_k(const float* __restrict__ flow,
                         const float* __restrict__ wt,
                         float* __restrict__ flowup) {
  int id = blockIdx.x * blockDim.x + threadIdx.x;
  if (id >= B * 2 * H * W) return;
  int x = id % W; int t = id / W;
  int y = t % H;  t /= H;
  int g = t % 2;  int b = t / 2;
  float acc = 0.f;
  for (int i = 0; i < 4; ++i) {
    int sy = y + i - 2;
    if (sy < 0 || (sy & 1)) continue;
    int p = sy >> 1; if (p >= Hh) continue;
    for (int j = 0; j < 4; ++j) {
      int sx = x + j - 2;
      if (sx < 0 || (sx & 1)) continue;
      int q = sx >> 1; if (q >= Wh) continue;
      acc += wt[g * 16 + (3 - i) * 4 + (3 - j)] *
             flow[((b * 2 + g) * Hh + p) * Wh + q];
    }
  }
  flowup[id] = acc;
}

// ---------------------------------------------------------------------------
// 2) Bilinear warp of feats[:,1] by flow_up (zero outside + coverage mask)
// ---------------------------------------------------------------------------
__global__ void warp_k(const float* __restrict__ feats,
                       const float* __restrict__ flowup,
                       float* __restrict__ warped) {
  int id = blockIdx.x * blockDim.x + threadIdx.x;
  if (id >= B * H * W) return;
  int x = id % W; int t = id / W;
  int y = t % H;  int b = t / H;

  float f0 = flowup[((b * 2 + 0) * H + y) * W + x];
  float f1 = flowup[((b * 2 + 1) * H + y) * W + x];
  float gx = -1.f + 2.f * x / (W - 1);
  float gy = -1.f + 2.f * y / (H - 1);
  float sx = gx + f0 * (2.0f / (W - 1) / DIV_FLOW);
  float sy = gy + f1 * (2.0f / (H - 1) / DIV_FLOW);
  float xx = (sx + 1.f) * (W - 1) * 0.5f;
  float yy = (sy + 1.f) * (H - 1) * 0.5f;
  float x0 = floorf(xx), y0 = floorf(yy);

  float cxi[4] = {x0, x0 + 1.f, x0, x0 + 1.f};
  float cyi[4] = {y0, y0, y0 + 1.f, y0 + 1.f};
  float cw[4]  = {(x0 + 1.f - xx) * (y0 + 1.f - yy),
                  (xx - x0)       * (y0 + 1.f - yy),
                  (x0 + 1.f - xx) * (yy - y0),
                  (xx - x0)       * (yy - y0)};
  int   idxs[4]; float wv[4]; float msk = 0.f;
  for (int k = 0; k < 4; ++k) {
    int xi = (int)cxi[k], yi = (int)cyi[k];
    float valid = (cxi[k] >= 0.f && cxi[k] < (float)W &&
                   cyi[k] >= 0.f && cyi[k] < (float)H) ? 1.f : 0.f;
    int xc = xi < 0 ? 0 : (xi > W - 1 ? W - 1 : xi);
    int yc = yi < 0 ? 0 : (yi > H - 1 ? H - 1 : yi);
    idxs[k] = yc * W + xc;
    wv[k]   = cw[k] * valid;
    msk    += wv[k];
  }
  float mask = (msk >= 1.f) ? 1.f : 0.f;

  const float* img = feats + ((size_t)(b * 2 + 1) * C) * (H * W);
  float* od = warped + (size_t)b * C * H * W + y * W + x;
  for (int c = 0; c < C; ++c) {
    const float* ic = img + (size_t)c * H * W;
    float v = wv[0] * ic[idxs[0]] + wv[1] * ic[idxs[1]] +
              wv[2] * ic[idxs[2]] + wv[3] * ic[idxs[3]];
    od[(size_t)c * H * W] = v * mask;
  }
}

// ---------------------------------------------------------------------------
// 3) 9x9 correlation -> lrelu -> /64 -> f16 padded-HWC buffer (96 ch, halo)
// ---------------------------------------------------------------------------
__global__ void corr_k(const float* __restrict__ feats,
                       const float* __restrict__ warped,
                       _Float16* __restrict__ corr) {
  int id = blockIdx.x * blockDim.x + threadIdx.x;
  if (id >= B * H * W) return;
  int x = id % W; int t = id / W;
  int y = t % H;  int b = t / H;

  const float* f1 = feats + ((size_t)(b * 2 + 0) * C) * (H * W) + y * W + x;
  const float* wb = warped + (size_t)b * C * H * W;
  _Float16* o = corr + ((size_t)(b * Hp + y + 1) * Wp + (x + 1)) * 96;

  for (int d = 0; d < 81; ++d) {
    int di = d / 9 - PR, dj = d % 9 - PR;
    int yy = y + di, xx = x + dj;
    float acc = 0.f;
    if (yy >= 0 && yy < H && xx >= 0 && xx < W) {
      const float* w2 = wb + yy * W + xx;
#pragma unroll 8
      for (int c = 0; c < C; ++c)
        acc += f1[(size_t)c * H * W] * w2[(size_t)c * H * W];
    }
    acc = (acc > 0.f ? acc : 0.1f * acc) * (1.0f / 64.0f);
    o[d] = (_Float16)acc;
  }
  for (int d = 81; d < 96; ++d) o[d] = (_Float16)0.f;  // channel padding
}

// ---------------------------------------------------------------------------
// Zero the 1-pixel halo of a padded HWC activation buffer (stride = Cp).
// ---------------------------------------------------------------------------
__global__ void zero_halo_k(_Float16* __restrict__ buf, int Cp) {
  int id = blockIdx.x * blockDim.x + threadIdx.x;
  if (id >= B * HALO * Cp) return;
  int c = id % Cp; int t = id / Cp;
  int p = t % HALO; int b = t / HALO;
  int yy, xx;
  if (p < Wp)            { yy = 0;      xx = p; }
  else if (p < 2 * Wp)   { yy = Hp - 1; xx = p - Wp; }
  else { int q = p - 2 * Wp; yy = 1 + (q >> 1); xx = (q & 1) ? (Wp - 1) : 0; }
  buf[((size_t)(b * Hp + yy) * Wp + xx) * Cp + c] = (_Float16)0.f;
}

// ---------------------------------------------------------------------------
// Pack conv weights (Cout,Cin,3,3) f32 -> f16 WMMA B-fragment layout:
//   [ktile][ntile][lane 0..31][16 halfs], lane l: n=l&15, K=(l>>4)*16 + j.
// ---------------------------------------------------------------------------
__global__ void pack_w_k(const float* __restrict__ w, _Float16* __restrict__ o,
                         int Cout, int Cin, int Cinp, int ntiles, int total) {
  int id = blockIdx.x * blockDim.x + threadIdx.x;
  if (id >= total) return;
  int j    = id & 15;
  int lane = (id >> 4) & 31;
  int nt   = (id >> 9) % ntiles;
  int kt   = id / (512 * ntiles);
  int cct  = Cinp / 32;
  int tap  = kt / cct;
  int cc   = kt % cct;
  int ky = tap / 3, kx = tap % 3;
  int n  = lane & 15;
  int k  = ((lane >> 4) * 16) + j;       // k_local 0..31
  int cin  = cc * 32 + k;
  int cout = nt * 16 + n;
  float v = 0.f;
  if (cin < Cin && cout < Cout)
    v = w[(((size_t)cout * Cin + cin) * 3 + ky) * 3 + kx];
  o[id] = (_Float16)v;
}

// ---------------------------------------------------------------------------
// Implicit-GEMM 3x3 conv. One workgroup = one 32-pixel tile (MT=2 x 16 px);
// NW waves split the cout tiles (NT each), every B fragment is reused across
// both pixel tiles. The 3x34xCin halo patch is DMA'd into LDS once per
// workgroup by the Tensor Data Mover (tensor_load_to_lds, TENSORcnt), then
// A fragments come from ds_load_b128. All strides are template constants so
// every load is base + immediate offset (schedulable, partial s_wait).
// ---------------------------------------------------------------------------
template <int NT, int CINP, int NW, int COUTP, bool FINAL>
__global__ __launch_bounds__(NW * 32) void conv_k(
    const _Float16* __restrict__ in, _Float16* __restrict__ outh,
    const _Float16* __restrict__ wpk, const float* __restrict__ bias,
    const float* __restrict__ flowup, float* __restrict__ outf) {
  constexpr int CCT  = CINP / 32;       // K tiles per tap
  constexpr int NTT  = COUTP / 16;      // total n tiles
  constexpr int COLS = 16 * MT + 2;     // staged columns (34)
  constexpr int ROWH = COLS * CINP;     // halfs per staged row
  __shared__ _Float16 smem[3 * ROWH];

  int tmp = blockIdx.x;
  int xt  = tmp % (W / (16 * MT)); tmp /= (W / (16 * MT));
  int y   = tmp % H; int b = tmp / H;
  int x0  = xt * (16 * MT);
  int lane = threadIdx.x & 31;
  int wv   = threadIdx.x >> 5;
  int grp  = lane >> 4, lm = lane & 15;

  // ---- Stage the (y..y+2) x (x0..x0+33) x CINP patch into LDS -------------
#if __has_builtin(__builtin_amdgcn_tensor_load_to_lds) && \
    __has_builtin(__builtin_amdgcn_s_wait_tensorcnt)
  if (threadIdx.x < 32) {
    const unsigned row_dw    = ROWH / 2;            // dwords per row
    const unsigned stride_dw = (unsigned)Wp * CINP / 2;
    unsigned long long ga =
        (unsigned long long)(uintptr_t)(in + ((size_t)(b * Hp + y) * Wp + x0) * CINP);
    unsigned lds_base = (unsigned)(uintptr_t)(&smem[0]);
    u32x4 g0 = {1u,                                   // count=1 (valid D#)
                lds_base,                             // D#.lds_addr
                (unsigned)ga,                         // D#.global_addr[31:0]
                (unsigned)(ga >> 32) | (2u << 30)};   // ga[56:32] | type=2
    i32x8 g1 = {(int)(2u << 16),                      // data_size = 4 bytes
                (int)((row_dw & 0xffffu) << 16),      // tensor_dim0[15:0]
                (int)(3u << 16),                      // tensor_dim1 = 3 rows
                (int)((row_dw & 0xffffu) << 16),      // tile_dim0 = row_dw
                (int)3,                               // tile_dim1 = 3
                (int)stride_dw,                       // tensor_dim0_stride
                0, 0};
    i32x4 gz = {0, 0, 0, 0};                          // <=2D: groups 2/3 null
#if __clang_major__ >= 23
    i32x8 gz8 = {0, 0, 0, 0, 0, 0, 0, 0};
    __builtin_amdgcn_tensor_load_to_lds(g0, g1, gz, gz, gz8, 0);
#else
    __builtin_amdgcn_tensor_load_to_lds(g0, g1, gz, gz, 0);
#endif
    __builtin_amdgcn_s_wait_tensorcnt(0);
  }
#else
  for (int i = threadIdx.x; i < 3 * ROWH / 8; i += NW * 32) {
    int ky = i / (ROWH / 8);
    int r  = i % (ROWH / 8);
    const _Float16* src =
        in + ((size_t)(b * Hp + y + ky) * Wp + x0) * CINP + r * 8;
    *(v8h*)(smem + ky * ROWH + r * 8) = *(const v8h*)src;
  }
#endif
  __syncthreads();

  __builtin_prefetch(wpk + (size_t)lane * 16, 0, 0);  // global_prefetch_b8

  // Per-lane constant base pointers: all loop offsets are immediates.
  const _Float16* abase = smem + lm * CINP + grp * 16;
  const _Float16* bbase = wpk + (size_t)(wv * NT) * 512 + lane * 16;

  v8f acc[NT][MT];
#pragma unroll
  for (int t = 0; t < NT; ++t)
#pragma unroll
    for (int mt = 0; mt < MT; ++mt)
#pragma unroll
      for (int e = 0; e < 8; ++e) acc[t][mt][e] = 0.f;

#pragma unroll
  for (int kt = 0; kt < 9 * CCT; ++kt) {
    const int tap = kt / CCT, cc = kt % CCT;
    const int ky = tap / 3, kx = tap % 3;
    v16h a[MT];
#pragma unroll
    for (int mt = 0; mt < MT; ++mt) {
      const _Float16* p0 =
          abase + ((ky * COLS + mt * 16 + kx) * CINP + cc * 32);
      v8h alo = *(const v8h*)p0;         // ds_load_b128, imm offset
      v8h ahi = *(const v8h*)(p0 + 8);  // ds_load_b128, imm offset
#pragma unroll
      for (int i = 0; i < 8; ++i) { a[mt][i] = alo[i]; a[mt][8 + i] = ahi[i]; }
    }
#pragma unroll
    for (int t = 0; t < NT; ++t) {
      const _Float16* wp = bbase + ((size_t)kt * NTT + t) * 512;
      v8h blo = *(const v8h*)wp;         // global_load_b128, imm offset
      v8h bhi = *(const v8h*)(wp + 8);
      v16h bb;
#pragma unroll
      for (int i = 0; i < 8; ++i) { bb[i] = blo[i]; bb[8 + i] = bhi[i]; }
#pragma unroll
      for (int mt = 0; mt < MT; ++mt)
        acc[t][mt] = __builtin_amdgcn_wmma_f32_16x16x32_f16(
            false, a[mt], false, bb, (short)0, acc[t][mt], false, false);
    }
  }

  if (FINAL) {  // conv6: cout 0..1 valid, fuse residual, f32 NCHW to d_out
    if (lm < 2) {
      float bv = bias[lm];
#pragma unroll
      for (int mt = 0; mt < MT; ++mt)
#pragma unroll
        for (int r = 0; r < 8; ++r) {
          int m = r + grp * 8;
          int idx = ((b * 2 + lm) * H + y) * W + (x0 + mt * 16 + m);
          outf[idx] = flowup[idx] + acc[0][mt][r] + bv;
        }
    }
  } else {      // lrelu(0.1) + bias, write f16 HWC interior of next buffer
#pragma unroll
    for (int t = 0; t < NT; ++t) {
      int cout = (wv * NT + t) * 16 + lm;
      float bv = bias[cout];
#pragma unroll
      for (int mt = 0; mt < MT; ++mt)
#pragma unroll
        for (int r = 0; r < 8; ++r) {
          int m = r + grp * 8;
          float v = acc[t][mt][r] + bv;
          v = v > 0.f ? v : 0.1f * v;
          outh[((size_t)(b * Hp + y + 1) * Wp + (x0 + mt * 16 + m + 1)) *
                   COUTP + cout] = (_Float16)v;
        }
    }
  }
}

// ---------------------------------------------------------------------------
extern "C" void kernel_launch(void* const* d_in, const int* in_sizes, int n_in,
                              void* d_out, int out_size, void* d_ws,
                              size_t ws_size, hipStream_t stream) {
  const float* feats = (const float*)d_in[0];
  const float* flow  = (const float*)d_in[1];
  const float* upw   = (const float*)d_in[2];
  const float* wl[6] = {(const float*)d_in[3],  (const float*)d_in[5],
                        (const float*)d_in[7],  (const float*)d_in[9],
                        (const float*)d_in[11], (const float*)d_in[13]};
  const float* bl[6] = {(const float*)d_in[4],  (const float*)d_in[6],
                        (const float*)d_in[8],  (const float*)d_in[10],
                        (const float*)d_in[12], (const float*)d_in[14]};
  float* out = (float*)d_out;

  char* ws = (char*)d_ws;
  size_t off = 0;
  auto alloc = [&](size_t bytes) -> char* {
    char* p = ws + off;
    off += (bytes + 255) & ~(size_t)255;
    return p;
  };
  float*    flowup = (float*)alloc(sizeof(float) * (size_t)B * 2 * H * W);
  float*    warped = (float*)alloc(sizeof(float) * (size_t)B * C * H * W);
  _Float16* corr   = (_Float16*)alloc(2ull * B * Hp * Wp * 96);
  _Float16* actX   = (_Float16*)alloc(2ull * B * Hp * Wp * 128);
  _Float16* actY   = (_Float16*)alloc(2ull * B * Hp * Wp * 128);
  _Float16* pk[6];
  const int ktl[6] = {27, 36, 36, 27, 18, 9};   // 9*Cinp/32
  const int ntl[6] = {8, 8, 6, 4, 2, 1};        // Coutp/16
  for (int i = 0; i < 6; ++i)
    pk[i] = (_Float16*)alloc(2ull * ktl[i] * ntl[i] * 512);

  const int Cout_[6] = {128, 128, 96, 64, 32, 2};
  const int Cin_[6]  = {81, 128, 128, 96, 64, 32};
  const int Cinp_[6] = {96, 128, 128, 96, 64, 32};
  for (int i = 0; i < 6; ++i) {
    int total = ktl[i] * ntl[i] * 512;
    pack_w_k<<<(total + 255) / 256, 256, 0, stream>>>(
        wl[i], pk[i], Cout_[i], Cin_[i], Cinp_[i], ntl[i], total);
  }

  upflow_k<<<(B * 2 * H * W + 255) / 256, 256, 0, stream>>>(flow, upw, flowup);
  warp_k<<<(B * H * W + 255) / 256, 256, 0, stream>>>(feats, flowup, warped);
  zero_halo_k<<<(B * HALO * 96 + 255) / 256, 256, 0, stream>>>(corr, 96);
  corr_k<<<(B * H * W + 255) / 256, 256, 0, stream>>>(feats, warped, corr);

  auto zh = [&](_Float16* buf, int cp) {
    zero_halo_k<<<(B * HALO * cp + 255) / 256, 256, 0, stream>>>(buf, cp);
  };
  dim3 cgrid(B * H * (W / (16 * MT)));  // one workgroup per 32-pixel tile

  zh(actX, 128);
  conv_k<2, 96, 4, 128, false><<<cgrid, 128, 0, stream>>>(
      corr, actX, pk[0], bl[0], nullptr, nullptr);
  zh(actY, 128);
  conv_k<2, 128, 4, 128, false><<<cgrid, 128, 0, stream>>>(
      actX, actY, pk[1], bl[1], nullptr, nullptr);
  zh(actX, 96);
  conv_k<2, 128, 3, 96, false><<<cgrid, 96, 0, stream>>>(
      actY, actX, pk[2], bl[2], nullptr, nullptr);
  zh(actY, 64);
  conv_k<2, 96, 2, 64, false><<<cgrid, 64, 0, stream>>>(
      actX, actY, pk[3], bl[3], nullptr, nullptr);
  zh(actX, 32);
  conv_k<2, 64, 1, 32, false><<<cgrid, 32, 0, stream>>>(
      actY, actX, pk[4], bl[4], nullptr, nullptr);
  conv_k<1, 32, 1, 16, true><<<cgrid, 32, 0, stream>>>(
      actX, nullptr, pk[5], bl[5], flowup, out);
}
